// NoisyTopKGate_79422535238243
// MI455X (gfx1250) — matrix operand order repaired and checked by reading, not compiled, not run on hardware
//
#include <hip/hip_runtime.h>
#include <cstdint>

// ---------------------------------------------------------------------------
// NoisyTopKGate for MI455X (gfx1250): fused dual-GEMM (h@[W_g|W_n]) with
// v_wmma_f32_16x16x32_bf16, double-buffered LDS staging (1 barrier / 128 K),
// wave32-parallel epilogue (softplus / softmax / shuffle-based top-8).
// ---------------------------------------------------------------------------

typedef __attribute__((ext_vector_type(16))) __bf16 v16bf;
typedef __attribute__((ext_vector_type(8)))  float  v8f;

#define N_TOK   65536
#define IN_DIM  4096
#define NEXP    64
#define NCOL    128        // [W_g | W_n] fused columns
#define TOPKK   8
#define ROWS    64         // rows of h per workgroup
#define KCHUNK  128        // K staged per barrier (4 WMMA sub-steps)
#define KSUB    4
#define NCHUNK  (IN_DIM / KCHUNK)
#define NWAVES  8
#define BLOCK   256

union Frag16 {             // one WMMA 16-bit operand: 16 bf16 = 32 B = 2x b128
  v16bf v;
  uint4 q[2];
};

__device__ __forceinline__ unsigned pk_bf16(float a, float b) {
  unsigned ua = __float_as_uint(a), ub = __float_as_uint(b);
  ua = (ua + 0x7FFFu + ((ua >> 16) & 1u)) >> 16;   // round-to-nearest-even
  ub = (ub + 0x7FFFu + ((ub >> 16) & 1u)) >> 16;
  return ua | (ub << 16);
}

// ---------------------------------------------------------------------------
// Kernel 1: pack W = [W_g | W_n] (f32 row-major [K, 64] each) into bf16 in the
// B-operand (32x16) lane order used here, k contiguous per lane half as in A:
//   Bws[((kt*8 + nt)*32 + lane)*16 + e] = W[kt*32 + (lane>>4)*16 + e]
//                                          [col = nt*16 + (lane&15)]
// 1 MB total -> L2-resident for the main kernel.
// ---------------------------------------------------------------------------
__global__ __launch_bounds__(256)
void pack_B_kernel(const float* __restrict__ Wg, const float* __restrict__ Wn,
                   unsigned short* __restrict__ Bws) {
  int i    = blockIdx.x * 256 + threadIdx.x;   // < 4096*128
  int e    =  i        & 15;
  int lane = (i >> 4)  & 31;
  int nt   = (i >> 9)  & 7;
  int kt   =  i >> 12;
  int k    = kt * 32 + ((lane >> 4) << 4) + e;
  int col  = nt * 16 + (lane & 15);
  float w  = (col < NEXP) ? Wg[(size_t)k * NEXP + col]
                          : Wn[(size_t)k * NEXP + (col - NEXP)];
  unsigned u = __float_as_uint(w);
  Bws[i] = (unsigned short)((u + 0x7FFFu + ((u >> 16) & 1u)) >> 16);
}

// ---------------------------------------------------------------------------
// Kernel 2: per WG, 64 rows x 128 logits.
// Wave w = N-tile w (16 cols); 4 M-tiles of 16 rows -> 4 accumulators.
// ---------------------------------------------------------------------------
__global__ __launch_bounds__(BLOCK)
void gate_kernel(const float* __restrict__ h,
                 const unsigned short* __restrict__ Bws,
                 const float* __restrict__ noise,
                 float* __restrict__ out_sparse,
                 int*   __restrict__ out_idx,
                 float* __restrict__ out_full) {
  __shared__ unsigned short ldsA[2][ROWS * KCHUNK];  // 2 x 16 KB bf16 A tiles
  __shared__ float          ldsC[ROWS * NCOL];       // 32 KB logits / gates

  const int tid  = threadIdx.x;
  const int lane = tid & 31;
  const int wave = tid >> 5;          // n-tile index 0..7
  const int hi   = lane >> 4;         // half-wave select
  const int lr   = lane & 15;
  const size_t rowBase = (size_t)blockIdx.x * ROWS;

  v8f acc[4] = {};                    // 4 M-tiles x (16x16 f32)

  // Staging: thread (wave,lane), j=0..7 loads h[row = j*8+wave][lane*4 ..+3]
  // of the current 128-wide K chunk. Fully coalesced float4 loads.
  const float* hbase = h + (rowBase + wave) * (size_t)IN_DIM + lane * 4;

  float4 st[8];
  #pragma unroll
  for (int j = 0; j < 8; ++j)
    st[j] = *(const float4*)(hbase + (size_t)(j * 8) * IN_DIM);

  // store chunk 0 into buffer 0
  #pragma unroll
  for (int j = 0; j < 8; ++j) {
    uint2 p;
    p.x = pk_bf16(st[j].x, st[j].y);
    p.y = pk_bf16(st[j].z, st[j].w);
    *(uint2*)&ldsA[0][(j * 8 + wave) * KCHUNK + lane * 4] = p;
  }
  __syncthreads();

  #pragma unroll 1
  for (int kc = 0; kc < NCHUNK; ++kc) {
    const unsigned short* cur = ldsA[kc & 1];
    unsigned short*       nxt = ldsA[(kc + 1) & 1];

    if (kc + 1 < NCHUNK) {            // global prefetch of next chunk
      #pragma unroll
      for (int j = 0; j < 8; ++j)
        st[j] = *(const float4*)(hbase + (size_t)(j * 8) * IN_DIM
                                        + (size_t)(kc + 1) * KCHUNK);
    }

    #pragma unroll
    for (int ks = 0; ks < KSUB; ++ks) {
      // B fragment for this 32-K sub-step: two coalesced global b128 (L2 hit)
      Frag16 b;
      const uint4* bp = (const uint4*)(Bws +
          (((size_t)(kc * KSUB + ks) * NWAVES + wave) * 32 + lane) * 16);
      b.q[0] = bp[0];
      b.q[1] = bp[1];

      #pragma unroll
      for (int mt = 0; mt < 4; ++mt) {
        // A 16x32 fragment, same k order as the B pack: lane(hi,lr) -> row mt*16+lr,
        // k [ks*32+hi*16 ..+15] contiguous
        Frag16 a;
        const uint4* ap = (const uint4*)&cur[(mt * 16 + lr) * KCHUNK];
        a.q[0] = ap[ks * 4 + 2 * hi];
        a.q[1] = ap[ks * 4 + 2 * hi + 1];
        acc[mt] = __builtin_amdgcn_wmma_f32_16x16x32_bf16(
            false, a.v, false, b.v, (short)0, acc[mt], false, false);
      }
    }

    if (kc + 1 < NCHUNK) {            // stage next chunk into the other buffer
      #pragma unroll
      for (int j = 0; j < 8; ++j) {
        uint2 p;
        p.x = pk_bf16(st[j].x, st[j].y);
        p.y = pk_bf16(st[j].z, st[j].w);
        *(uint2*)&nxt[(j * 8 + wave) * KCHUNK + lane * 4] = p;
      }
    }
    __syncthreads();                  // one barrier per 128-K chunk
  }

  // ---- dump accumulators to LDS: C/D layout row = 8*hi + r, col = lr ----
  #pragma unroll
  for (int mt = 0; mt < 4; ++mt) {
    #pragma unroll
    for (int r = 0; r < 8; ++r) {
      int row = mt * 16 + hi * 8 + r;
      ldsC[row * NCOL + wave * 16 + lr] = acc[mt][r];
    }
  }
  __syncthreads();

  // ---- wave-parallel epilogue: wave w owns rows w*8 .. w*8+7 ----
  #pragma unroll 1
  for (int rr = 0; rr < 8; ++rr) {
    const int    rl  = wave * 8 + rr;
    const size_t row = rowBase + rl;
    const float* crow = &ldsC[rl * NCOL];

    float c0 = crow[lane];            // clean logits, experts lane / lane+32
    float c1 = crow[lane + 32];
    float n0 = crow[lane + 64];       // noise logits
    float n1 = crow[lane + 96];
    float z0 = noise[row * NEXP + lane];
    float z1 = noise[row * NEXP + lane + 32];

    float sp0 = (n0 > 0.f) ? n0 + log1pf(__expf(-n0)) : log1pf(__expf(n0));
    float sp1 = (n1 > 0.f) ? n1 + log1pf(__expf(-n1)) : log1pf(__expf(n1));
    float q0 = c0 + z0 * (sp0 + 0.01f);
    float q1 = c1 + z1 * (sp1 + 0.01f);

    // softmax over 64 experts: 5-step wave32 butterflies
    float m = fmaxf(q0, q1);
    #pragma unroll
    for (int off = 16; off > 0; off >>= 1) m = fmaxf(m, __shfl_xor(m, off, 32));
    float g0 = __expf(q0 - m), g1 = __expf(q1 - m);
    float s = g0 + g1;
    #pragma unroll
    for (int off = 16; off > 0; off >>= 1) s += __shfl_xor(s, off, 32);
    float inv = 1.f / s;
    g0 *= inv; g1 *= inv;

    float* full = out_full   + row * NEXP;
    float* spo  = out_sparse + row * NEXP;
    full[lane] = g0;  full[lane + 32] = g1;
    spo[lane]  = 0.f; spo[lane + 32]  = 0.f;   // overwritten below for top-k

    // top-8: argmax butterflies with lower-index tie-break (jax.lax.top_k)
    float a0 = g0, a1 = g1;
    float myv = 0.f, tv0 = 0.f;
    int   myi = 0;
    #pragma unroll
    for (int kk = 0; kk < TOPKK; ++kk) {
      float v  = (a0 >= a1) ? a0   : a1;        // prefer lower expert index
      int   ei = (a0 >= a1) ? lane : lane + 32;
      #pragma unroll
      for (int off = 16; off > 0; off >>= 1) {
        float ov = __shfl_xor(v, off, 32);
        int   oi = __shfl_xor(ei, off, 32);
        if (ov > v || (ov == v && oi < ei)) { v = ov; ei = oi; }
      }
      // v, ei now uniform across the wave
      if (kk == 0)   tv0 = v;
      if (lane == kk) { myv = v; myi = ei; }
      if (ei == lane)      a0 = -1.f;           // mark taken
      if (ei == lane + 32) a1 = -1.f;
    }

    // renormalize over the 8 winners (held by lanes 0..7)
    float w  = (lane < TOPKK) ? __expf(myv - tv0) : 0.f;
    float s2 = w;
    #pragma unroll
    for (int off = 16; off > 0; off >>= 1) s2 += __shfl_xor(s2, off, 32);

    if (lane < TOPKK) {
      out_idx[row * TOPKK + lane] = myi;
      spo[myi] = w / s2;     // same-wave stores ordered: overwrites the zero
    }
  }
}

extern "C" void kernel_launch(void* const* d_in, const int* in_sizes, int n_in,
                              void* d_out, int out_size, void* d_ws, size_t ws_size,
                              hipStream_t stream) {
  const float* h  = (const float*)d_in[0];
  const float* Wg = (const float*)d_in[1];
  const float* Wn = (const float*)d_in[2];
  const float* nz = (const float*)d_in[3];

  float* out        = (float*)d_out;
  float* out_sparse = out;                                               // [N,64]
  int*   out_idx    = (int*)(out + (size_t)N_TOK * NEXP);                // [N,8]
  float* out_full   = out + (size_t)N_TOK * NEXP + (size_t)N_TOK * TOPKK; // [N,64]

  unsigned short* Bws = (unsigned short*)d_ws;   // 4096*128 bf16 = 1 MB

  pack_B_kernel<<<(IN_DIM * NCOL) / 256, 256, 0, stream>>>(Wg, Wn, Bws);
  gate_kernel<<<N_TOK / ROWS, BLOCK, 0, stream>>>(h, Bws, nz,
                                                  out_sparse, out_idx, out_full);
}
